// RuchbahStableMoEGate_4131758538903
// MI455X (gfx1250) — matrix operand order repaired and checked by reading, not compiled, not run on hardware
//
#include <hip/hip_runtime.h>

// ---------------------------------------------------------------------------
// MoE gate: logits = x[T,1024] @ W[64,1024]^T, softmax(64), top-2,
// softmax over the 2 selected scores. bf16x3-split WMMA for fp32-class
// accuracy while streaming x at HBM rate (memory-bound: ~128MB / 23.3TB/s).
// ---------------------------------------------------------------------------

typedef __attribute__((ext_vector_type(16))) __bf16 v16bf;
typedef __attribute__((ext_vector_type(8)))  float  v8f;

#define GATE_H   1024   // hidden dim
#define GATE_E   64     // experts
#define GATE_KCH 32     // K per WMMA chunk
#define WAVES_PER_BLOCK 8
#define TOK_PER_BLOCK   (WAVES_PER_BLOCK * 16)   // 128 tokens / block

// ---- prep: split W into bf16 hi/lo (residual) arrays in workspace --------
__global__ void gate_split_w_kernel(const float* __restrict__ W,
                                    __bf16* __restrict__ whi,
                                    __bf16* __restrict__ wlo, int n) {
  int i = blockIdx.x * blockDim.x + threadIdx.x;
  if (i < n) {
    float f = W[i];
    __bf16 h = (__bf16)f;             // RNE
    whi[i] = h;
    wlo[i] = (__bf16)(f - (float)h);  // residual
  }
}

// Load 16 contiguous bf16 (32 B) into a v16bf B-fragment slice.
static __device__ inline v16bf load_bfrag(const __bf16* __restrict__ p) {
  union { uint4 q[2]; v16bf v; } u;
  const uint4* s = (const uint4*)p;
  u.q[0] = s[0];
  u.q[1] = s[1];
  return u.v;
}

__global__ __launch_bounds__(256) void
RuchbahStableMoEGate_kernel(const float* __restrict__ x,
                            const __bf16* __restrict__ whi,
                            const __bf16* __restrict__ wlo,
                            float* __restrict__ d_out, int T) {
  __shared__ float sl[WAVES_PER_BLOCK][16][GATE_E];   // 32 KB logits staging

  const int tid  = threadIdx.x;
  const int lane = tid & 31;
  const int wave = tid >> 5;
  const int h    = lane >> 4;   // lane half (K-half selector)
  const int n    = lane & 15;   // row within half / B column

  const int tileTok = blockIdx.x * TOK_PER_BLOCK + wave * 16;  // 16 tokens/wave

  v8f acc[4];
#pragma unroll
  for (int g = 0; g < 4; ++g) {
    v8f z = {0.f, 0.f, 0.f, 0.f, 0.f, 0.f, 0.f, 0.f};
    acc[g] = z;
  }

  // Base pointer for this lane's A row (row = tileTok + n).
  const float* __restrict__ xrow = x + (size_t)(tileTok + n) * GATE_H;

#pragma unroll 1
  for (int k0 = 0; k0 < GATE_H; k0 += GATE_KCH) {
    // ---- 1) issue the long-latency HBM loads for the A fragment first ----
    // A fragment (contiguous K order, matching B): elements 0..15 = K
    // h*16..h*16+15 of this chunk. Per chunk the wave
    // consumes exactly one full 128B line per row.
    const float* xr = xrow + k0 + h * 16;
    float4 f0 = *(const float4*)(xr);
    float4 f1 = *(const float4*)(xr + 4);
    float4 f2 = *(const float4*)(xr + 8);
    float4 f3 = *(const float4*)(xr + 12);
    __builtin_prefetch(xr + 2 * GATE_KCH, 0, 1);   // 2 chunks (256 B) ahead

    // ---- 2) issue ALL cache-resident B loads for the chunk in one burst --
    v16bf bh[4], bl[4];
#pragma unroll
    for (int g = 0; g < 4; ++g) {
      const size_t woff = (size_t)(g * 16 + n) * GATE_H + k0 + h * 16;
      bh[g] = load_bfrag(whi + woff);
      bl[g] = load_bfrag(wlo + woff);
    }

    // ---- 3) fp32 -> bf16 hi/lo split (depends only on A; overlaps B tail)
    float v[16] = {f0.x, f0.y, f0.z, f0.w, f1.x, f1.y, f1.z, f1.w,
                   f2.x, f2.y, f2.z, f2.w, f3.x, f3.y, f3.z, f3.w};
    v16bf ahi, alo;
#pragma unroll
    for (int i = 0; i < 16; ++i) {
      __bf16 hh = (__bf16)v[i];
      ahi[i] = hh;
      alo[i] = (__bf16)(v[i] - (float)hh);
    }

    // ---- 4) 12 back-to-back WMMAs (3 split terms x 4 expert groups) ------
#pragma unroll
    for (int g = 0; g < 4; ++g) {
      acc[g] = __builtin_amdgcn_wmma_f32_16x16x32_bf16(
          false, ahi, false, bh[g], (short)0, acc[g], false, false);
      acc[g] = __builtin_amdgcn_wmma_f32_16x16x32_bf16(
          false, alo, false, bh[g], (short)0, acc[g], false, false);
      acc[g] = __builtin_amdgcn_wmma_f32_16x16x32_bf16(
          false, ahi, false, bl[g], (short)0, acc[g], false, false);
    }
  }

  // ---- stage logits to LDS (C layout: VGPR j, lane l -> M=j+8*(l>>4), N=l&15)
#pragma unroll
  for (int g = 0; g < 4; ++g) {
#pragma unroll
    for (int j = 0; j < 8; ++j) {
      sl[wave][j + 8 * h][g * 16 + n] = acc[g][j];
    }
  }
  __syncthreads();

  // ---- per-token softmax + top-2 + renorm softmax of scores -------------
  if (tid < TOK_PER_BLOCK) {
    const int tok = blockIdx.x * TOK_PER_BLOCK + tid;
    const float* lrow = &sl[tid >> 4][tid & 15][0];

    float v1 = -3.4e38f, v2 = -3.4e38f;
    int   i1 = 0, i2 = 0;
#pragma unroll
    for (int e = 0; e < GATE_E; ++e) {
      float l = lrow[e];
      if (l > v1) { v2 = v1; i2 = i1; v1 = l; i1 = e; }
      else if (l > v2) { v2 = l; i2 = e; }
    }
    float sum = 0.f;
#pragma unroll
    for (int e = 0; e < GATE_E; ++e) sum += __expf(lrow[e] - v1);

    // full-softmax scores of the two winners
    float inv = 1.f / sum;
    float s1 = inv;                       // exp(0)/sum
    float s2 = __expf(v2 - v1) * inv;
    // renormalize: softmax([s1, s2]) in fp32 (s1 >= s2)
    float r  = __expf(s2 - s1);
    float p1 = 1.f / (1.f + r);
    float p2 = r / (1.f + r);

    float* outs = d_out;                          // [T,2] scores
    int*   outi = (int*)(d_out + 2 * (size_t)T);  // [T,2] indices
    outs[2 * tok + 0] = p1;
    outs[2 * tok + 1] = p2;
    outi[2 * tok + 0] = i1;
    outi[2 * tok + 1] = i2;
    if (tok == 0) d_out[4 * (size_t)T] = 0.f;     // trailing scalar zero
  }
}

extern "C" void kernel_launch(void* const* d_in, const int* in_sizes, int n_in,
                              void* d_out, int out_size, void* d_ws, size_t ws_size,
                              hipStream_t stream) {
  const float* x = (const float*)d_in[0];   // [4,8192,1024] fp32
  const float* W = (const float*)d_in[1];   // [64,1024] fp32

  const int T  = in_sizes[0] / GATE_H;      // 32768 tokens
  const int wn = in_sizes[1];               // 65536 = 64*1024

  __bf16* whi = (__bf16*)d_ws;              // 128 KB
  __bf16* wlo = whi + wn;                   // 128 KB (total 256 KB scratch)

  gate_split_w_kernel<<<(wn + 255) / 256, 256, 0, stream>>>(W, whi, wlo, wn);

  RuchbahStableMoEGate_kernel<<<T / TOK_PER_BLOCK, 256, 0, stream>>>(
      x, whi, wlo, (float*)d_out, T);
}
